// Conv2dB_61160334295585
// MI455X (gfx1250) — matrix operand order repaired and checked by reading, not compiled, not run on hardware
//
#include <hip/hip_runtime.h>
#include <math.h>

// ---------------- problem constants ----------------
#define B_IMG 16
#define CIN   128
#define COUT  128
#define HW    56
#define L_    (HW*HW)        // 3136 spatial positions
#define KTOT  (9*CIN)        // 1152 = reduction dim
#define NKB   (KTOT/32)      // 36 K-blocks of 32
#define NTOT  256            // 2*COUT fused output channels (cos | sin)
#define MWG   128            // M rows per workgroup
#define KB_CHUNK (2*16*32*16) // bf16 elems per K-block in packed B (hi+lo) = 32KB

typedef __attribute__((ext_vector_type(16))) __bf16 v16bf;
typedef __attribute__((ext_vector_type(8)))  float  v8f;
typedef __attribute__((ext_vector_type(4)))  unsigned int v4u;
typedef __attribute__((ext_vector_type(8)))  int v8i;
typedef __attribute__((ext_vector_type(4)))  int v4i;

__device__ __forceinline__ unsigned short bf16_rne(float f) {
  unsigned int u = __float_as_uint(f);
  u += 0x7FFFu + ((u >> 16) & 1u);
  return (unsigned short)(u >> 16);
}

// ---------------------------------------------------------------------------
// Kernel 1: fold cos/sin(b_k) into conv weights, split fp32 -> bf16 hi/lo,
// pack into per-lane WMMA B-fragment layout:
//   Bp[kb][pre][ntile][lane][e], 16 bf16 per lane (32B contiguous).
//   B 32x16 bf16, K order used here: lane<16 -> K=e,    N=lane
//                        lane>=16 -> K=e+16, N=lane-16
// ---------------------------------------------------------------------------
__global__ void prep_weights(const float* __restrict__ w,
                             const float* __restrict__ bk,
                             unsigned short* __restrict__ Bp) {
  int idx = blockIdx.x * 256 + threadIdx.x;
  if (idx >= NKB * 16 * 32 * 16) return;
  int e    = idx & 15;
  int lane = (idx >> 4) & 31;
  int nt   = (idx >> 9) & 15;
  int kb   = idx >> 13;

  int Klocal = (lane < 16) ? e : (e + 16);
  int Kg   = kb * 32 + Klocal;
  int kidx = Kg >> 7;        // 0..8 : 3x3 tap
  int c    = Kg & 127;       // input channel
  int Ng   = nt * 16 + (lane & 15);
  int o    = Ng & 127;

  float wv  = w[(kidx * COUT + o) * CIN + c];
  float ang = bk[kidx * COUT + o];
  float f   = (Ng < 128) ? (wv * cosf(ang)) : (wv * sinf(ang));

  unsigned short hi = bf16_rne(f);
  float hif = __uint_as_float(((unsigned int)hi) << 16);
  unsigned short lo = bf16_rne(f - hif);

  size_t base = (size_t)kb * KB_CHUNK;
  Bp[base + ((size_t)(0 * 16 + nt) * 32 + lane) * 16 + e] = hi;
  Bp[base + ((size_t)(1 * 16 + nt) * 32 + lane) * 16 + e] = lo;
}

// phase epilogue math (matches reference arctan/peak logic)
__device__ __forceinline__ float2 phase_eval(float a, float bb, float bv) {
  const float PI  = 3.14159265358979323846f;
  const float EPS = 1e-5f;
  bool ag = a > 0.0f, bg = bb > 0.0f;
  float th, peak;
  if (ag == bg) {
    th   = atanf(bb / (a + EPS));
    peak = ag ? (0.5f * PI) : (-0.5f * PI);
  } else {
    th   = -atanf(a / (bb + EPS));
    peak = bg ? 0.0f : -PI;
  }
  float theta = peak - th;
  float s, c;
  __sincosf(theta, &s, &c);
  return make_float2(s * a + c * bb + bv, theta * (1.0f / PI));
}

// ---------------------------------------------------------------------------
// Kernel 2: implicit-GEMM conv (M=50176, N=256, K=1152) with bf16x3 WMMA,
// TDM async weight staging, fused phase epilogue.
// 256 threads = 8 waves, 4(M) x 2(N); wave tile 32 rows x 8 paired n-tiles.
// ---------------------------------------------------------------------------
__global__ __launch_bounds__(256, 1) void conv_wmma(
    const float* __restrict__ x,
    const unsigned short* __restrict__ Bp,
    const float* __restrict__ bias,
    float* __restrict__ out) {

  __shared__ __align__(16) unsigned short Ahi_s[MWG * 32];
  __shared__ __align__(16) unsigned short Alo_s[MWG * 32];
  __shared__ __align__(16) unsigned short Bs[KB_CHUNK];

  const int tid  = threadIdx.x;
  const int lane = tid & 31;
  const int wav  = tid >> 5;
  const int wy   = wav >> 1;   // 0..3  (M group)
  const int wx   = wav & 1;    // 0..1  (N group)
  const int m0   = blockIdx.x * MWG;

  v8f acc[2][8];
  const v8f vzero = {0.f,0.f,0.f,0.f,0.f,0.f,0.f,0.f};
  #pragma unroll
  for (int i = 0; i < 2; ++i)
    #pragma unroll
    for (int j = 0; j < 8; ++j) acc[i][j] = vzero;

  // im2col staging coordinates: thread t -> row (t&127), c-half (t>>7)
  const int mloc  = tid & 127;
  const int chalf = tid >> 7;
  const int m     = m0 + mloc;
  const int bimg  = m / L_;
  const int lpos  = m % L_;
  const int ypos  = lpos / HW;
  const int xpos  = lpos % HW;

  union Frag { uint4 q[2]; v16bf v; };

#if defined(__has_builtin) && __has_builtin(__builtin_amdgcn_tensor_load_to_lds)
  // Loop-invariant pieces of the Tensor DMA descriptor (D#):
  // 1-D copy, data_size=8B, tile_dim0 = tensor_dim0 = 4096 units = 32KB.
  const unsigned lds_b = (unsigned)(unsigned long long)(uintptr_t)&Bs[0];
  const v8i g1 = { (int)(3u << 16),          // data_size=3 (8B), wg_mask=0
                   (int)(4096u << 16),       // tensor_dim0[15:0]
                   (int)(1u << 16),          // tensor_dim0[31:16]=0, tensor_dim1=1
                   (int)(4096u << 16),       // tile_dim0=4096
                   (int)1,                   // tile_dim1=1, tile_dim2=0
                   (int)4096,                // tensor_dim0_stride[31:0]
                   0, 0 };
  const v4i g2 = { 1, 1, 0, 0 };             // tensor_dim2=1, tensor_dim3=1
  const v4i g3 = { 0, (int)(1u << 16), 0, 0 }; // tensor_dim4=1, tile_dim4=0
  const v8i gz = { 0, 0, 0, 0, 0, 0, 0, 0 }; // extra group (clang-23 6-arg form)
#endif

  for (int kb = 0; kb < NKB; ++kb) {
#if defined(__has_builtin) && __has_builtin(__builtin_amdgcn_tensor_load_to_lds)
    // ---- stage B via Tensor Data Mover: one 32KB DMA, wave 0 only ----
    if (wav == 0) {
      unsigned long long ga =
          (unsigned long long)(uintptr_t)(Bp + (size_t)kb * KB_CHUNK);
      v4u g0 = { 1u,                               // count=1, user mode
                 lds_b,                            // lds_addr
                 (unsigned)(ga & 0xFFFFFFFFu),     // global_addr[31:0]
                 (unsigned)((ga >> 32) & 0x1FFFFFFu) | (2u << 30) }; // type=2
      __builtin_amdgcn_tensor_load_to_lds(g0, g1, g2, g3, gz, 0);
    }
#else
    // ---- fallback: plain 32KB copy (already frag-packed) ----
    {
      const uint4* src = (const uint4*)(Bp + (size_t)kb * KB_CHUNK);
      uint4* dst = (uint4*)Bs;
      #pragma unroll
      for (int i = 0; i < 8; ++i) dst[tid + 256 * i] = src[tid + 256 * i];
    }
#endif

    // ---- stage A (im2col gather, fp32 -> bf16 hi/lo, K contiguous) ----
    const int kidx = kb >> 2;
    const int dy = kidx / 3 - 1;
    const int dx = kidx % 3 - 1;
    const int c0 = (kb & 3) * 32;
    const int ya = ypos + dy, xa = xpos + dx;
    const bool valid = (ya >= 0) && (ya < HW) && (xa >= 0) && (xa < HW);
    const float* xb = x + (size_t)bimg * CIN * L_ + (size_t)ya * HW + xa;
    #pragma unroll
    for (int cc = 0; cc < 16; ++cc) {
      int c = chalf * 16 + cc;
      float v = valid ? xb[(size_t)(c0 + c) * L_] : 0.0f;
      unsigned short hi = bf16_rne(v);
      float hif = __uint_as_float(((unsigned int)hi) << 16);
      unsigned short lo = bf16_rne(v - hif);
      // K stored contiguously: lane half h reads K = 16h..16h+15
      int p = c;
      Ahi_s[mloc * 32 + p] = hi;
      Alo_s[mloc * 32 + p] = lo;
    }

#if defined(__has_builtin) && __has_builtin(__builtin_amdgcn_tensor_load_to_lds)
    if (wav == 0) __builtin_amdgcn_s_wait_tensorcnt(0);
#endif
    __syncthreads();

    // ---- A fragments for this wave's 2 M-subtiles ----
    Frag ah[2], al[2];
    #pragma unroll
    for (int ms = 0; ms < 2; ++ms) {
      int row = wy * 32 + ms * 16 + (lane & 15);
      int off = row * 32 + (lane >> 4) * 16;
      ah[ms].q[0] = *(const uint4*)(Ahi_s + off);
      ah[ms].q[1] = *(const uint4*)(Ahi_s + off + 8);
      al[ms].q[0] = *(const uint4*)(Alo_s + off);
      al[ms].q[1] = *(const uint4*)(Alo_s + off + 8);
    }

    // ---- 8 n-tiles (paired cos/sin), 3 WMMAs each (bf16x3) ----
    #pragma unroll
    for (int j = 0; j < 8; ++j) {
      int t = (j < 4) ? (wx * 4 + j) : (8 + wx * 4 + (j - 4));
      Frag bh, bl;
      int bo0 = ((0 * 16 + t) * 32 + lane) * 16;
      bh.q[0] = *(const uint4*)(Bs + bo0);
      bh.q[1] = *(const uint4*)(Bs + bo0 + 8);
      int bo1 = ((1 * 16 + t) * 32 + lane) * 16;
      bl.q[0] = *(const uint4*)(Bs + bo1);
      bl.q[1] = *(const uint4*)(Bs + bo1 + 8);
      #pragma unroll
      for (int ms = 0; ms < 2; ++ms) {
        acc[ms][j] = __builtin_amdgcn_wmma_f32_16x16x32_bf16(
            false, ah[ms].v, false, bh.v, (short)0, acc[ms][j], false, false);
        acc[ms][j] = __builtin_amdgcn_wmma_f32_16x16x32_bf16(
            false, al[ms].v, false, bh.v, (short)0, acc[ms][j], false, false);
        acc[ms][j] = __builtin_amdgcn_wmma_f32_16x16x32_bf16(
            false, ah[ms].v, false, bl.v, (short)0, acc[ms][j], false, false);
      }
    }
    __syncthreads();
  }

  // ---- phase epilogue: theta from (a, bb), write conv_out and theta/pi ----
  const int nloc = lane & 15;
  const int half = lane >> 4;

  #pragma unroll
  for (int ms = 0; ms < 2; ++ms) {
    #pragma unroll
    for (int j = 0; j < 4; ++j) {
      int ch = (wx * 4 + j) * 16 + nloc;   // 0..127
      float bv = bias[ch];
      v8f av  = acc[ms][j];      // cos branch (a)
      v8f bbv = acc[ms][j + 4];  // sin branch (bb)
      int mb = m0 + wy * 32 + ms * 16 + half * 8;
      int b  = mb / L_;
      int l  = mb % L_;          // 8-aligned, never crosses an image

      float2 r0 = phase_eval(av[0], bbv[0], bv);
      float2 r1 = phase_eval(av[1], bbv[1], bv);
      float2 r2 = phase_eval(av[2], bbv[2], bv);
      float2 r3 = phase_eval(av[3], bbv[3], bv);
      float2 r4 = phase_eval(av[4], bbv[4], bv);
      float2 r5 = phase_eval(av[5], bbv[5], bv);
      float2 r6 = phase_eval(av[6], bbv[6], bv);
      float2 r7 = phase_eval(av[7], bbv[7], bv);

      float4* p0 = (float4*)(out + ((size_t)b * NTOT + ch) * L_ + l);
      p0[0] = make_float4(r0.x, r1.x, r2.x, r3.x);
      p0[1] = make_float4(r4.x, r5.x, r6.x, r7.x);
      float4* p1 = (float4*)(out + ((size_t)b * NTOT + 128 + ch) * L_ + l);
      p1[0] = make_float4(r0.y, r1.y, r2.y, r3.y);
      p1[1] = make_float4(r4.y, r5.y, r6.y, r7.y);
    }
  }
}

// ---------------------------------------------------------------------------
extern "C" void kernel_launch(void* const* d_in, const int* in_sizes, int n_in,
                              void* d_out, int out_size, void* d_ws, size_t ws_size,
                              hipStream_t stream) {
  const float* x    = (const float*)d_in[0];   // (16,128,56,56)
  const float* w    = (const float*)d_in[1];   // (1,9,128,128)
  const float* bk   = (const float*)d_in[2];   // (1,9,1,128,1)
  const float* bias = (const float*)d_in[3];   // (128,)
  unsigned short* Bp = (unsigned short*)d_ws;  // needs 36*16384*2 = 1.13 MB

  // pack fused weights (hi/lo bf16, WMMA B-frag layout)
  prep_weights<<<(NKB * 16 * 32 * 16 + 255) / 256, 256, 0, stream>>>(w, bk, Bp);
  // implicit-GEMM conv + epilogue: 50176/128 = 392 blocks
  conv_wmma<<<(B_IMG * L_) / MWG, 256, 0, stream>>>(x, Bp, bias, (float*)d_out);
}
